// ShiftConv2D0_27865747816604
// MI455X (gfx1250) — matrix operand order repaired and checked by reading, not compiled, not run on hardware
//
#include <hip/hip_runtime.h>
#include <hip/hip_bf16.h>

// ShiftConv2D: masked 3x3 conv == per-channel-group shift + [262144 x 180] @ [180 x 180] GEMM.
// bf16x3 split-precision WMMA (v_wmma_f32_16x16x32_bf16), fp32 accumulate.

typedef __attribute__((ext_vector_type(16))) __bf16 v16bf;
typedef __attribute__((ext_vector_type(8)))  __bf16 v8bf;
typedef __attribute__((ext_vector_type(8)))  float  v8f;

#define N_  16
#define H_  128
#define W_  128
#define C_  180
#define F_  180
#define KP  192   // K padded to multiple of 32
#define FP  192   // F padded to multiple of 16
#define MT  32    // pixels (M rows) per workgroup

// ---------------------------------------------------------------------------
// Pre-pass: Keff[c][f] = kernel[tap(c)][c][f]  ->  column-major bf16 hi/lo
// Layout: B[f*KP + k]  (k contiguous) so a WMMA B-fragment is one contiguous
// 32-byte load per lane. Zero-padded to 192x192.
// ---------------------------------------------------------------------------
__global__ __launch_bounds__(256) void build_Bmat(const float* __restrict__ kernel,
                                                  __bf16* __restrict__ Bhi,
                                                  __bf16* __restrict__ Blo) {
  int idx = blockIdx.x * 256 + threadIdx.x;
  if (idx >= FP * KP) return;
  int f = idx / KP, k = idx % KP;
  float w = 0.f;
  if (f < F_ && k < C_) {
    int g = k / 36;
    // taps: g0->(1,2) right, g1->(1,0) left, g2->(2,1) down, g3->(0,1) up, g4->(1,1) center
    int ky = (g == 2) ? 2 : ((g == 3) ? 0 : 1);
    int kx = (g == 0) ? 2 : ((g == 1) ? 0 : 1);
    w = kernel[((ky * 3 + kx) * C_ + k) * F_ + f];
  }
  __bf16 h = (__bf16)w;
  __bf16 l = (__bf16)(w - (float)h);
  Bhi[idx] = h;
  Blo[idx] = l;
}

// ---------------------------------------------------------------------------
// Main kernel: one workgroup = 32 consecutive pixels of one W-row x all 192 F.
// 8 waves: wave = (mi, fq); mi in {0,1} picks 16-pixel block, fq covers 3
// F-blocks of 16. K-loop: 6 chunks of 32, 3 WMMAs each (bf16x3).
// ---------------------------------------------------------------------------
__global__ __launch_bounds__(256) void shiftconv_wmma(const float* __restrict__ in,
                                                      const __bf16* __restrict__ Bhi,
                                                      const __bf16* __restrict__ Blo,
                                                      const float* __restrict__ bias,
                                                      float* __restrict__ out) {
  __shared__ __align__(16) __bf16 Ahi[MT][KP];
  __shared__ __align__(16) __bf16 Alo[MT][KP];

  const int tid = threadIdx.x;
  const int pixbase = blockIdx.x * MT;

  // ---- Stage A tile: shift-gather + fp32 -> bf16 hi/lo split into LDS ----
  // Channels are innermost (NHWC), consecutive tid -> consecutive c: coalesced.
  for (int s = tid; s < MT * KP; s += 256) {
    int p = s / KP, c = s % KP;
    int P   = pixbase + p;
    int n   = P / (H_ * W_);
    int rem = P % (H_ * W_);
    int y = rem / W_;
    int x = rem % W_;
    float v = 0.f;
    if (c < C_) {
      int g  = c / 36;
      int yy = y + (g == 2) - (g == 3);
      int xx = x + (g == 0) - (g == 1);
      if (yy >= 0 && yy < H_ && xx >= 0 && xx < W_)
        v = in[((n * H_ + yy) * W_ + xx) * C_ + c];
    }
    __bf16 h = (__bf16)v;
    Ahi[p][c] = h;
    Alo[p][c] = (__bf16)(v - (float)h);
  }
  __syncthreads();

  const int wave = tid >> 5;
  const int lane = tid & 31;
  const int half = lane >> 4;   // 0: lanes 0-15, 1: lanes 16-31
  const int lrow = lane & 15;
  const int mi   = wave >> 2;   // which 16-pixel block
  const int fq   = wave & 3;    // covers F-blocks fq*3 .. fq*3+2

  v8f acc0 = {}, acc1 = {}, acc2 = {};

  const int m = mi * 16 + lrow; // A row for this lane (M = lane%16 per ISA layout)

  for (int kc = 0; kc < KP / 32; ++kc) {
    const int kb = kc * 32;
    // A-fragment 16x32 bf16: lane half 0 -> K 0-15,
    // lane half 1 -> K 16-31 (matches B); each 8-element run is one 16B LDS load.
    v8bf ah0 = *(const v8bf*)&Ahi[m][kb + half * 16];
    v8bf ah1 = *(const v8bf*)&Ahi[m][kb + half * 16 + 8];
    v8bf al0 = *(const v8bf*)&Alo[m][kb + half * 16];
    v8bf al1 = *(const v8bf*)&Alo[m][kb + half * 16 + 8];
    v16bf ah, al;
#pragma unroll
    for (int j = 0; j < 8; ++j) {
      ah[j] = ah0[j]; ah[8 + j] = ah1[j];
      al[j] = al0[j]; al[8 + j] = al1[j];
    }

#pragma unroll
    for (int t = 0; t < 3; ++t) {
      const int fb = fq * 3 + t;
      const int fcol = fb * 16 + lrow;
      // B-fragment 32x16 bf16: lane's column = lane%16; lanes 0-15 hold
      // K kb..kb+15, lanes 16-31 hold K kb+16..kb+31 -> one contiguous 32B load.
      v16bf bh = *(const v16bf*)&Bhi[fcol * KP + kb + half * 16];
      v16bf bl = *(const v16bf*)&Blo[fcol * KP + kb + half * 16];
      v8f& acc = (t == 0) ? acc0 : (t == 1 ? acc1 : acc2);
      // bf16x3: a*w ~= ah*wh + ah*wl + al*wh  (al*wl ~ 2^-16, dropped)
      acc = __builtin_amdgcn_wmma_f32_16x16x32_bf16(false, ah, false, bh, (short)0, acc, false, false);
      acc = __builtin_amdgcn_wmma_f32_16x16x32_bf16(false, ah, false, bl, (short)0, acc, false, false);
      acc = __builtin_amdgcn_wmma_f32_16x16x32_bf16(false, al, false, bh, (short)0, acc, false, false);
    }
  }

  // ---- Epilogue: C/D f32 16x16 layout: col = lane%16, row = v + 8*(lane>=16)
#pragma unroll
  for (int t = 0; t < 3; ++t) {
    const int fb = fq * 3 + t;
    const int fcol = fb * 16 + lrow;
    if (fcol < F_) {
      const float b = bias[fcol];
      const v8f& acc = (t == 0) ? acc0 : (t == 1 ? acc1 : acc2);
#pragma unroll
      for (int v = 0; v < 8; ++v) {
        const int row = v + half * 8;
        const int P = pixbase + mi * 16 + row;
        out[P * F_ + fcol] = acc[v] + b;
      }
    }
  }
}

extern "C" void kernel_launch(void* const* d_in, const int* in_sizes, int n_in,
                              void* d_out, int out_size, void* d_ws, size_t ws_size,
                              hipStream_t stream) {
  const float* inp    = (const float*)d_in[0];   // (16,128,128,180) f32
  const float* kernel = (const float*)d_in[1];   // (3,3,180,180)    f32
  const float* bias   = (const float*)d_in[2];   // (1,1,1,180)      f32
  float* out = (float*)d_out;                    // (16,128,128,180) f32

  // Workspace: Bhi + Blo, each 192*192 bf16 (total 147456 bytes).
  __bf16* Bhi = (__bf16*)d_ws;
  __bf16* Blo = Bhi + FP * KP;

  build_Bmat<<<(FP * KP + 255) / 256, 256, 0, stream>>>(kernel, Bhi, Blo);

  const int nblocks = (N_ * H_ * W_) / MT; // 8192
  shiftconv_wmma<<<nblocks, 256, 0, stream>>>(inp, Bhi, Blo, bias, out);
}
